// VirtualTryOnModel_74277164417321
// MI455X (gfx1250) — matrix operand order repaired and checked by reading, not compiled, not run on hardware
//
#include <hip/hip_runtime.h>

// ---------------------------------------------------------------------------
// MI455X (gfx1250) implementation, round 4.
// - Channels-last, zero-haloed, channel-padded f16 activations.
// - Conv/deconv as implicit GEMM on v_wmma_f32_16x16x32_f16 (wave32).
// - Register tiling: each wave computes a (MT*16) x 32 output tile
//   (MT=2 normally): 4 accumulators, A reused across pixel tiles, B reused
//   across cout tiles -> 2 b128 loads per wmma.
// - Tap-outer / channel-inner K loop: per tap one base pointer, inner loads
//   are contiguous b128 with immediate offsets.
// - Deconv per output-parity class (fixed 2x2 tap set) with per-class
//   pre-packed weights.
// ---------------------------------------------------------------------------

typedef __attribute__((ext_vector_type(16))) _Float16 v16h;
typedef __attribute__((ext_vector_type(8)))  _Float16 v8h;
typedef __attribute__((ext_vector_type(8)))  float    v8f;

#define TPS_P 18

static __device__ __forceinline__ v16h cat16(v8h a, v8h b) {
  return __builtin_shufflevector(a, b, 0, 1, 2, 3, 4, 5, 6, 7,
                                       8, 9, 10, 11, 12, 13, 14, 15);
}

static __device__ __forceinline__ v8f wmma_step(v16h A, v16h B, v8f C) {
  return __builtin_amdgcn_wmma_f32_16x16x32_f16(false, A, false, B, (short)0,
                                                C, false, false);
}

static __device__ __forceinline__ float apply_act(float v, int act) {
  if (act == 1) v = fmaxf(v, 0.f);
  else if (act == 2) v = 1.f / (1.f + __expf(-v));
  else if (act == 3) v = tanhf(v);
  return v;
}

// ---------------------------------------------------------------------------
// utility fills
// ---------------------------------------------------------------------------
__global__ void __launch_bounds__(256)
fill_zero_f16(_Float16* __restrict__ p, int n) {
  int i = blockIdx.x * blockDim.x + threadIdx.x;
  if (i < n) p[i] = (_Float16)0.f;
}

// zero the 1-pixel spatial halo of a [B, Hp, Wp, C] buffer
__global__ void __launch_bounds__(256)
zero_halo(_Float16* __restrict__ p, int Bn, int Hp, int Wp, int C) {
  int hc = 2 * Wp + 2 * (Hp - 2);
  int total = Bn * hc * C;
  int idx = blockIdx.x * blockDim.x + threadIdx.x;
  if (idx >= total) return;
  int b = idx / (hc * C);
  int rem = idx - b * hc * C;
  int cell = rem / C;
  int c = rem - cell * C;
  int y, x;
  if (cell < Wp)          { y = 0;      x = cell; }
  else if (cell < 2 * Wp) { y = Hp - 1; x = cell - Wp; }
  else {
    int s = cell - 2 * Wp;
    y = 1 + (s >> 1);
    x = (s & 1) ? (Wp - 1) : 0;
  }
  p[(((size_t)b * Hp + y) * Wp + x) * C + c] = (_Float16)0.f;
}

// ---------------------------------------------------------------------------
// weight packing
// ---------------------------------------------------------------------------
__global__ void __launch_bounds__(256)
pack_conv_w(const float* __restrict__ w, _Float16* __restrict__ out,
            int Cout, int Cin, int cinShift) {
  int Cinp = 1 << cinShift;
  int K = 16 * Cinp;
  int idx = blockIdx.x * blockDim.x + threadIdx.x;
  if (idx >= Cout * K) return;
  int m = idx / K;
  int kk = idx - m * K;
  int tap = kk >> cinShift;
  int ci = kk & (Cinp - 1);
  float v = 0.f;
  if (ci < Cin) v = w[(m * Cin + ci) * 16 + tap];
  out[idx] = (_Float16)v;
}

__global__ void __launch_bounds__(256)
pack_deconv_w(const float* __restrict__ w, _Float16* __restrict__ out,
              int Cin, int Cout, int CoutPad) {
  int K = Cin * 4;
  int total = 4 * CoutPad * K;
  int idx = blockIdx.x * blockDim.x + threadIdx.x;
  if (idx >= total) return;
  int cls = idx / (CoutPad * K);
  int rem = idx - cls * CoutPad * K;
  int m = rem / K;
  int kk = rem - m * K;
  int t = kk / Cin;
  int ci = kk - t * Cin;
  int py = cls >> 1, px = cls & 1;
  int ky = ((py + 1) & 1) + ((t >> 1) << 1);
  int kx = ((px + 1) & 1) + ((t & 1) << 1);
  float v = 0.f;
  if (m < Cout) v = w[((ci * Cout + m) * 4 + ky) * 4 + kx];
  out[idx] = (_Float16)v;
}

// ---------------------------------------------------------------------------
// TPS 18x18 solve per batch (Gaussian elimination, one lane)
// ---------------------------------------------------------------------------
__global__ void __launch_bounds__(32)
tps_solve(const float* __restrict__ kp, const float* __restrict__ cp,
          float* __restrict__ wcoef) {
  if (threadIdx.x != 0) return;
  int b = blockIdx.x;
  float A[TPS_P][TPS_P];
  float rhs[TPS_P][2];
  const float* d = kp + b * TPS_P * 2;
  for (int i = 0; i < TPS_P; ++i) {
    for (int j = 0; j < TPS_P; ++j) {
      float dx = d[i * 2 + 0] - d[j * 2 + 0];
      float dy = d[i * 2 + 1] - d[j * 2 + 1];
      float r2 = dx * dx + dy * dy;
      float ph = (r2 > 0.f) ? 0.5f * r2 * __logf(r2) : 0.f; // r^2*ln(r)
      A[i][j] = ph - ((i == j) ? 1e-5f : 0.f);
    }
    rhs[i][0] = cp[i * 2 + 0];
    rhs[i][1] = cp[i * 2 + 1];
  }
  for (int c = 0; c < TPS_P; ++c) {
    int piv = c; float mx = fabsf(A[c][c]);
    for (int r = c + 1; r < TPS_P; ++r) {
      float v = fabsf(A[r][c]);
      if (v > mx) { mx = v; piv = r; }
    }
    if (piv != c) {
      for (int j = 0; j < TPS_P; ++j) { float t = A[c][j]; A[c][j] = A[piv][j]; A[piv][j] = t; }
      float t0 = rhs[c][0]; rhs[c][0] = rhs[piv][0]; rhs[piv][0] = t0;
      float t1 = rhs[c][1]; rhs[c][1] = rhs[piv][1]; rhs[piv][1] = t1;
    }
    float inv = 1.0f / A[c][c];
    for (int r = c + 1; r < TPS_P; ++r) {
      float f = A[r][c] * inv;
      for (int j = c; j < TPS_P; ++j) A[r][j] -= f * A[c][j];
      rhs[r][0] -= f * rhs[c][0];
      rhs[r][1] -= f * rhs[c][1];
    }
  }
  float* wc = wcoef + b * TPS_P * 2;
  for (int r = TPS_P - 1; r >= 0; --r) {
    float s0 = rhs[r][0], s1 = rhs[r][1];
    for (int j = r + 1; j < TPS_P; ++j) {
      s0 -= A[r][j] * wc[j * 2 + 0];
      s1 -= A[r][j] * wc[j * 2 + 1];
    }
    wc[r * 2 + 0] = s0 / A[r][r];
    wc[r * 2 + 1] = s1 / A[r][r];
  }
}

// ---------------------------------------------------------------------------
// body image -> channels 0..2 of xbuf (8ch) and ybuf (8ch), padded interiors
// ---------------------------------------------------------------------------
__global__ void __launch_bounds__(256)
body_to_xy(const float* __restrict__ body, _Float16* __restrict__ xbuf,
           _Float16* __restrict__ ybuf, int Bn, int H, int W) {
  int npix = H * W;
  int idx = blockIdx.x * blockDim.x + threadIdx.x;
  if (idx >= Bn * 3 * npix) return;
  int b = idx / (3 * npix);
  int rem = idx - b * 3 * npix;
  int c = rem / npix;
  int p = rem - c * npix;
  int gy = p / W;
  int gx = p - gy * W;
  int Hp = H + 2, Wp = W + 2;
  size_t o = (((size_t)b * Hp + gy + 1) * Wp + gx + 1) * 8 + c;
  _Float16 v = (_Float16)body[idx];
  xbuf[o] = v;
  ybuf[o] = v;
}

// ---------------------------------------------------------------------------
// TPS warp + bilinear sample -> channels 3..5 of xbuf and ybuf
// ---------------------------------------------------------------------------
__global__ void __launch_bounds__(256)
tps_warp(const float* __restrict__ cloth, const float* __restrict__ kp,
         const float* __restrict__ wcoef, _Float16* __restrict__ xbuf,
         _Float16* __restrict__ ybuf, int Bn, int H, int W) {
  int npix = H * W;
  int idx = blockIdx.x * blockDim.x + threadIdx.x;
  if (idx >= Bn * npix) return;
  int b = idx / npix;
  int p = idx - b * npix;
  int gy = p / W;
  int gx = p - gy * W;
  const float* d  = kp + b * TPS_P * 2;
  const float* wc = wcoef + b * TPS_P * 2;
  float fx = (float)gx, fy = (float)gy;
  float wx = 0.f, wy = 0.f;
  for (int i = 0; i < TPS_P; ++i) {
    float dx = fx - d[i * 2 + 0];
    float dy = fy - d[i * 2 + 1];
    float r2 = dx * dx + dy * dy;
    float ph = (r2 > 0.f) ? 0.5f * r2 * __logf(r2) : 0.f;
    wx += ph * wc[i * 2 + 0];
    wy += ph * wc[i * 2 + 1];
  }
  wx = fminf(fmaxf(wx, 0.f), (float)(W - 1));
  wy = fminf(fmaxf(wy, 0.f), (float)(H - 1));
  int x0 = (int)floorf(wx); x0 = x0 < 0 ? 0 : (x0 > W - 1 ? W - 1 : x0);
  int y0 = (int)floorf(wy); y0 = y0 < 0 ? 0 : (y0 > H - 1 ? H - 1 : y0);
  int x1 = (x0 + 1 < W - 1) ? x0 + 1 : W - 1;
  int y1 = (y0 + 1 < H - 1) ? y0 + 1 : H - 1;
  float x0f = (float)x0, x1f = (float)x1, y0f = (float)y0, y1f = (float)y1;
  float wa = (x1f - wx) * (y1f - wy);
  float wb = (x1f - wx) * (wy - y0f);
  float wcw = (wx - x0f) * (y1f - wy);
  float wd = (wx - x0f) * (wy - y0f);
  int Hp = H + 2, Wp = W + 2;
  const float* cb = cloth + (size_t)b * 3 * npix;
  size_t obase = (((size_t)b * Hp + gy + 1) * Wp + gx + 1) * 8 + 3;
  for (int c = 0; c < 3; ++c) {
    const float* img = cb + (size_t)c * npix;
    float v = wa * img[y0 * W + x0] + wb * img[y1 * W + x0] +
              wcw * img[y0 * W + x1] + wd * img[y1 * W + x1];
    _Float16 hv = (_Float16)v;
    xbuf[obase + c] = hv;
    ybuf[obase + c] = hv;
  }
}

// ---------------------------------------------------------------------------
// store helper (padded channels-last f16 or NCHW f32)
// ---------------------------------------------------------------------------
static __device__ __forceinline__ void
store_tile(const v8f& acc, const float* __restrict__ bias,
           _Float16* __restrict__ out16, float* __restrict__ out32, int b,
           int Cout, int mo0, int oy, int ox, int Hpo, int Wpo, int Cpo,
           int chOff, int act, int np, int pixLinear) {
  if (out16) {
    long obase = (((long)b * Hpo + oy + 1) * Wpo + ox + 1) * Cpo + chOff;
    if (mo0 + 8 <= Cout) {
      v8h ov;
#pragma unroll
      for (int r = 0; r < 8; ++r)
        ov[r] = (_Float16)apply_act(acc[r] + bias[mo0 + r], act);
      *(v8h*)(out16 + obase + mo0) = ov;
    } else {
#pragma unroll
      for (int r = 0; r < 8; ++r) {
        int mo = mo0 + r;
        if (mo < Cout)
          out16[obase + mo] = (_Float16)apply_act(acc[r] + bias[mo], act);
      }
    }
  } else {
#pragma unroll
    for (int r = 0; r < 8; ++r) {
      int mo = mo0 + r;
      if (mo < Cout)
        out32[((long)(b * Cout + mo)) * np + pixLinear] =
            apply_act(acc[r] + bias[mo], act);
    }
  }
}

// ---------------------------------------------------------------------------
// conv 4x4 s2 p1 : implicit GEMM on WMMA, (MT*16) x 32 output tile per wave.
// Input: padded channels-last [B,Hpin,Wpin,Cinp=1<<CS]; weights packed
// [Cout][16*Cinp].
// ---------------------------------------------------------------------------
template <int CS, int MT>
__global__ void __launch_bounds__(32)
conv4x4s2_wmma(const _Float16* __restrict__ in, const _Float16* __restrict__ wp,
               const float* __restrict__ bias, _Float16* __restrict__ out16,
               float* __restrict__ out32, int Hpin, int Wpin, int Cout,
               int Hout, int Wout, int Hpo, int Wpo, int Cpo, int chOff,
               int act) {
  constexpr int Cinp = 1 << CS;
  constexpr int K = 16 * Cinp;
  const int lane = threadIdx.x;
  const int lhi = lane >> 4;
  const int l15 = lane & 15;
  const int b = blockIdx.z;
  const int coutBase = blockIdx.y * (MT * 16);
  const int chunkLo = lhi << 3;

  const int pix0 = (blockIdx.x << 5) + l15;   // tile counts exact
  const int pix1 = pix0 + 16;
  const int oy0 = pix0 / Wout, ox0 = pix0 - oy0 * Wout;
  const int oy1 = pix1 / Wout, ox1 = pix1 - oy1 * Wout;
  const _Float16* __restrict__ base0 =
      in + ((((long)b * Hpin + 2 * oy0) * Wpin + 2 * ox0) << CS) + chunkLo;
  const _Float16* __restrict__ base1 =
      in + ((((long)b * Hpin + 2 * oy1) * Wpin + 2 * ox1) << CS) + chunkLo;
  const _Float16* __restrict__ arow0 =
      wp + (size_t)(coutBase + l15) * K + chunkLo;
  const _Float16* __restrict__ arow1 = arow0 + (size_t)16 * K;

  v8f acc00 = {}, acc01 = {}, acc10 = {}, acc11 = {};

  if constexpr (Cinp >= 32) {
#pragma unroll
    for (int tap = 0; tap < 16; ++tap) {
      const int toff = ((tap >> 2) * Wpin + (tap & 3)) << CS;
      const _Float16* bb0 = base0 + toff;
      const _Float16* bb1 = base1 + toff;
      const _Float16* aa0 = arow0 + tap * Cinp;
      const _Float16* aa1 = arow1 + tap * Cinp;
#pragma unroll
      for (int ci = 0; ci < Cinp; ci += 32) {
        v16h A0 = cat16(*(const v8h*)(aa0 + ci), *(const v8h*)(aa0 + ci + 16));
        v16h B0 = cat16(*(const v8h*)(bb0 + ci), *(const v8h*)(bb0 + ci + 16));
        v16h B1 = cat16(*(const v8h*)(bb1 + ci), *(const v8h*)(bb1 + ci + 16));
        acc00 = wmma_step(A0, B0, acc00);
        acc01 = wmma_step(A0, B1, acc01);
        if constexpr (MT == 2) {
          v16h A1 =
              cat16(*(const v8h*)(aa1 + ci), *(const v8h*)(aa1 + ci + 16));
          acc10 = wmma_step(A1, B0, acc10);
          acc11 = wmma_step(A1, B1, acc11);
        }
      }
    }
  } else {
    // Cinp == 8: each 32-wide K step spans 4 taps; chunk ci is always 0.
#pragma unroll
    for (int ks = 0; ks < K; ks += 32) {
      const int tA = (ks + chunkLo) >> 3;
      const int tB = (ks + chunkLo + 16) >> 3;
      const int offA = (((tA >> 2) * Wpin + (tA & 3)) << 3) - chunkLo;
      const int offB = (((tB >> 2) * Wpin + (tB & 3)) << 3) - chunkLo;
      v16h A0 = cat16(*(const v8h*)(arow0 + ks), *(const v8h*)(arow0 + ks + 16));
      v16h B0 = cat16(*(const v8h*)(base0 + offA), *(const v8h*)(base0 + offB));
      v16h B1 = cat16(*(const v8h*)(base1 + offA), *(const v8h*)(base1 + offB));
      acc00 = wmma_step(A0, B0, acc00);
      acc01 = wmma_step(A0, B1, acc01);
      if constexpr (MT == 2) {
        v16h A1 =
            cat16(*(const v8h*)(arow1 + ks), *(const v8h*)(arow1 + ks + 16));
        acc10 = wmma_step(A1, B0, acc10);
        acc11 = wmma_step(A1, B1, acc11);
      }
    }
  }

  const int np = Hout * Wout;
  store_tile(acc00, bias, out16, out32, b, Cout, coutBase + chunkLo, oy0, ox0,
             Hpo, Wpo, Cpo, chOff, act, np, pix0);
  store_tile(acc01, bias, out16, out32, b, Cout, coutBase + chunkLo, oy1, ox1,
             Hpo, Wpo, Cpo, chOff, act, np, pix1);
  if constexpr (MT == 2) {
    store_tile(acc10, bias, out16, out32, b, Cout, coutBase + 16 + chunkLo,
               oy0, ox0, Hpo, Wpo, Cpo, chOff, act, np, pix0);
    store_tile(acc11, bias, out16, out32, b, Cout, coutBase + 16 + chunkLo,
               oy1, ox1, Hpo, Wpo, Cpo, chOff, act, np, pix1);
  }
}

// ---------------------------------------------------------------------------
// deconv 4x4 s2 p1 : per-parity-class implicit GEMM on WMMA,
// (MT*16) x 32 output tile per wave. Weights per-class packed
// [4][CoutPad][4*Cin]. blockIdx.z encodes (batch, parity class).
// ---------------------------------------------------------------------------
template <int CS, int MT>
__global__ void __launch_bounds__(32)
deconv4x4s2_wmma(const _Float16* __restrict__ in, const _Float16* __restrict__ wp,
                 const float* __restrict__ bias, _Float16* __restrict__ out16,
                 float* __restrict__ out32, int Hpin, int Wpin, int Cout,
                 int CoutPad, int Hin, int Win, int Hpo, int Wpo, int Cpo,
                 int chOff, int act) {
  constexpr int Cin = 1 << CS;
  constexpr int K = 4 * Cin;
  const int lane = threadIdx.x;
  const int lhi = lane >> 4;
  const int l15 = lane & 15;
  const int bz = blockIdx.z;
  const int b = bz >> 2;
  const int cls = bz & 3;
  const int py = cls >> 1, px = cls & 1;
  const int coutBase = blockIdx.y * (MT * 16);
  const int chunkLo = lhi << 3;
  const int Wout = Win * 2;
  const int ky0 = (py + 1) & 1;
  const int kx0 = (px + 1) & 1;
  const long brow = (long)b * Hpin;

  const int n0 = (blockIdx.x << 5) + l15;
  const int n1 = n0 + 16;
  const int Yc0 = n0 / Win, Xc0 = n0 - Yc0 * Win;
  const int Yc1 = n1 / Win, Xc1 = n1 - Yc1 * Win;
  const int oy0 = 2 * Yc0 + py, ox0 = 2 * Xc0 + px;
  const int oy1 = 2 * Yc1 + py, ox1 = 2 * Xc1 + px;
  // padded input coords for the two tap parities (arith shift handles -1)
  const int iypA0 = ((oy0 + 1 - ky0) >> 1) + 1;
  const int iypB0 = ((oy0 - 1 - ky0) >> 1) + 1;
  const int ixpA0 = ((ox0 + 1 - kx0) >> 1) + 1;
  const int ixpB0 = ((ox0 - 1 - kx0) >> 1) + 1;
  const int iypA1 = ((oy1 + 1 - ky0) >> 1) + 1;
  const int iypB1 = ((oy1 - 1 - ky0) >> 1) + 1;
  const int ixpA1 = ((ox1 + 1 - kx0) >> 1) + 1;
  const int ixpB1 = ((ox1 - 1 - kx0) >> 1) + 1;

  const _Float16* __restrict__ arow0 =
      wp + ((size_t)cls * CoutPad + coutBase + l15) * K + chunkLo;
  const _Float16* __restrict__ arow1 = arow0 + (size_t)16 * K;

  v8f acc00 = {}, acc01 = {}, acc10 = {}, acc11 = {};
#pragma unroll
  for (int t = 0; t < 4; ++t) {
    const int y0 = (t & 2) ? iypB0 : iypA0;
    const int x0 = (t & 1) ? ixpB0 : ixpA0;
    const int y1 = (t & 2) ? iypB1 : iypA1;
    const int x1 = (t & 1) ? ixpB1 : ixpA1;
    const _Float16* bb0 = in + ((((brow + y0) * Wpin) + x0) << CS) + chunkLo;
    const _Float16* bb1 = in + ((((brow + y1) * Wpin) + x1) << CS) + chunkLo;
    const _Float16* aa0 = arow0 + t * Cin;
    const _Float16* aa1 = arow1 + t * Cin;
#pragma unroll
    for (int ci = 0; ci < Cin; ci += 32) {
      v16h A0 = cat16(*(const v8h*)(aa0 + ci), *(const v8h*)(aa0 + ci + 16));
      v16h B0 = cat16(*(const v8h*)(bb0 + ci), *(const v8h*)(bb0 + ci + 16));
      v16h B1 = cat16(*(const v8h*)(bb1 + ci), *(const v8h*)(bb1 + ci + 16));
      acc00 = wmma_step(A0, B0, acc00);
      acc01 = wmma_step(A0, B1, acc01);
      if constexpr (MT == 2) {
        v16h A1 = cat16(*(const v8h*)(aa1 + ci), *(const v8h*)(aa1 + ci + 16));
        acc10 = wmma_step(A1, B0, acc10);
        acc11 = wmma_step(A1, B1, acc11);
      }
    }
  }

  const int np = 4 * Hin * Win;
  store_tile(acc00, bias, out16, out32, b, Cout, coutBase + chunkLo, oy0, ox0,
             Hpo, Wpo, Cpo, chOff, act, np, oy0 * Wout + ox0);
  store_tile(acc01, bias, out16, out32, b, Cout, coutBase + chunkLo, oy1, ox1,
             Hpo, Wpo, Cpo, chOff, act, np, oy1 * Wout + ox1);
  if constexpr (MT == 2) {
    store_tile(acc10, bias, out16, out32, b, Cout, coutBase + 16 + chunkLo,
               oy0, ox0, Hpo, Wpo, Cpo, chOff, act, np, oy0 * Wout + ox0);
    store_tile(acc11, bias, out16, out32, b, Cout, coutBase + 16 + chunkLo,
               oy1, ox1, Hpo, Wpo, Cpo, chOff, act, np, oy1 * Wout + ox1);
  }
}

// ---------------------------------------------------------------------------
// Host-side orchestration
// ---------------------------------------------------------------------------
extern "C" void kernel_launch(void* const* d_in, const int* in_sizes, int n_in,
                              void* d_out, int out_size, void* d_ws, size_t ws_size,
                              hipStream_t stream) {
  (void)in_sizes; (void)n_in; (void)out_size; (void)ws_size;
  const int B = 16, H = 256, W = 192;
  const int npix = H * W;

  const float* cloth = (const float*)d_in[0];
  const float* body  = (const float*)d_in[1];
  const float* kp    = (const float*)d_in[2];
  const float* cp    = (const float*)d_in[3];
  const float* e1_w = (const float*)d_in[4];  const float* e1_b = (const float*)d_in[5];
  const float* e2_w = (const float*)d_in[6];  const float* e2_b = (const float*)d_in[7];
  const float* e3_w = (const float*)d_in[8];  const float* e3_b = (const float*)d_in[9];
  const float* sd1_w = (const float*)d_in[10]; const float* sd1_b = (const float*)d_in[11];
  const float* sd2_w = (const float*)d_in[12]; const float* sd2_b = (const float*)d_in[13];
  const float* sd3_w = (const float*)d_in[14]; const float* sd3_b = (const float*)d_in[15];
  const float* c1_w = (const float*)d_in[16]; const float* c1_b = (const float*)d_in[17];
  const float* c2_w = (const float*)d_in[18]; const float* c2_b = (const float*)d_in[19];
  const float* cd1_w = (const float*)d_in[20]; const float* cd1_b = (const float*)d_in[21];
  const float* cd2_w = (const float*)d_in[22]; const float* cd2_b = (const float*)d_in[23];

  char* ws = (char*)d_ws;
  size_t off = 0;
  auto alloc = [&](size_t bytes) -> void* {
    off = (off + 255) & ~(size_t)255;
    void* p = ws + off;
    off += bytes;
    return p;
  };
  float* wcoef = (float*)alloc(B * TPS_P * 2 * sizeof(float));
  _Float16* we1  = (_Float16*)alloc((size_t)64 * 128 * 2);        // Cinp=8
  _Float16* we2  = (_Float16*)alloc((size_t)128 * 1024 * 2);      // Cinp=64
  _Float16* we3  = (_Float16*)alloc((size_t)256 * 2048 * 2);      // Cinp=128
  _Float16* wsd1 = (_Float16*)alloc((size_t)4 * 128 * 1024 * 2);  // Cin=256
  _Float16* wsd2 = (_Float16*)alloc((size_t)4 * 64 * 512 * 2);    // Cin=128
  _Float16* wsd3 = (_Float16*)alloc((size_t)4 * 16 * 256 * 2);    // Cin=64, CoutPad=16
  _Float16* wc1  = (_Float16*)alloc((size_t)64 * 128 * 2);
  _Float16* wc2  = (_Float16*)alloc((size_t)128 * 1024 * 2);
  _Float16* wcd1 = (_Float16*)alloc((size_t)4 * 64 * 512 * 2);
  _Float16* wcd2 = (_Float16*)alloc((size_t)4 * 16 * 256 * 2);    // Cin=64, CoutPad=16
  const int xElems = B * 258 * 194 * 8;
  _Float16* xbuf = (_Float16*)alloc((size_t)xElems * 2);
  _Float16* ybuf = (_Float16*)alloc((size_t)xElems * 2);
  const size_t pElems = (size_t)B * 130 * 98 * 64;
  _Float16* ping = (_Float16*)alloc(pElems * 2);
  _Float16* pong = (_Float16*)alloc(pElems * 2);

  auto gsz = [](int n) { return (n + 255) / 256; };

  // 0) clear concat buffers (covers halo + padded channels)
  fill_zero_f16<<<gsz(xElems), 256, 0, stream>>>(xbuf, xElems);
  fill_zero_f16<<<gsz(xElems), 256, 0, stream>>>(ybuf, xElems);

  // 1) pack weights
  pack_conv_w<<<gsz(64 * 128), 256, 0, stream>>>(e1_w, we1, 64, 6, 3);
  pack_conv_w<<<gsz(128 * 1024), 256, 0, stream>>>(e2_w, we2, 128, 64, 6);
  pack_conv_w<<<gsz(256 * 2048), 256, 0, stream>>>(e3_w, we3, 256, 128, 7);
  pack_conv_w<<<gsz(64 * 128), 256, 0, stream>>>(c1_w, wc1, 64, 7, 3);
  pack_conv_w<<<gsz(128 * 1024), 256, 0, stream>>>(c2_w, wc2, 128, 64, 6);
  pack_deconv_w<<<gsz(4 * 128 * 1024), 256, 0, stream>>>(sd1_w, wsd1, 256, 128, 128);
  pack_deconv_w<<<gsz(4 * 64 * 512), 256, 0, stream>>>(sd2_w, wsd2, 128, 64, 64);
  pack_deconv_w<<<gsz(4 * 16 * 256), 256, 0, stream>>>(sd3_w, wsd3, 64, 1, 16);
  pack_deconv_w<<<gsz(4 * 64 * 512), 256, 0, stream>>>(cd1_w, wcd1, 128, 64, 64);
  pack_deconv_w<<<gsz(4 * 16 * 256), 256, 0, stream>>>(cd2_w, wcd2, 64, 3, 16);

  // 2) TPS + concat-buffer population
  tps_solve<<<B, 32, 0, stream>>>(kp, cp, wcoef);
  body_to_xy<<<gsz(B * 3 * npix), 256, 0, stream>>>(body, xbuf, ybuf, B, H, W);
  tps_warp<<<gsz(B * npix), 256, 0, stream>>>(cloth, kp, wcoef, xbuf, ybuf, B, H, W);

  auto halo = [&](_Float16* p, int Hp, int Wp, int C) {
    int n = B * (2 * Wp + 2 * (Hp - 2)) * C;
    zero_halo<<<gsz(n), 256, 0, stream>>>(p, B, Hp, Wp, C);
  };

  // 3) SegNet
  {
    dim3 g((128 * 96) / 32, 64 / 32, B);
    conv4x4s2_wmma<3, 2><<<g, 32, 0, stream>>>(xbuf, we1, e1_b, ping, nullptr,
        258, 194, 64, 128, 96, 130, 98, 64, 0, 1);
  }
  halo(ping, 130, 98, 64);
  {
    dim3 g((64 * 48) / 32, 128 / 32, B);
    conv4x4s2_wmma<6, 2><<<g, 32, 0, stream>>>(ping, we2, e2_b, pong, nullptr,
        130, 98, 128, 64, 48, 66, 50, 128, 0, 1);
  }
  halo(pong, 66, 50, 128);
  {
    dim3 g((32 * 24) / 32, 256 / 32, B);
    conv4x4s2_wmma<7, 2><<<g, 32, 0, stream>>>(pong, we3, e3_b, ping, nullptr,
        66, 50, 256, 32, 24, 34, 26, 256, 0, 1);
  }
  halo(ping, 34, 26, 256);
  {
    dim3 g((32 * 24) / 32, 128 / 32, B * 4);
    deconv4x4s2_wmma<8, 2><<<g, 32, 0, stream>>>(ping, wsd1, sd1_b, pong, nullptr,
        34, 26, 128, 128, 32, 24, 66, 50, 128, 0, 1);
  }
  halo(pong, 66, 50, 128);
  {
    dim3 g((64 * 48) / 32, 64 / 32, B * 4);
    deconv4x4s2_wmma<7, 2><<<g, 32, 0, stream>>>(pong, wsd2, sd2_b, ping, nullptr,
        66, 50, 64, 64, 64, 48, 130, 98, 64, 0, 1);
  }
  halo(ping, 130, 98, 64);
  {
    dim3 g((128 * 96) / 32, 1, B * 4);
    deconv4x4s2_wmma<6, 1><<<g, 32, 0, stream>>>(ping, wsd3, sd3_b, ybuf, nullptr,
        130, 98, 1, 16, 128, 96, 258, 194, 8, 6, 2);
  }

  // 4) CompNet
  {
    dim3 g((128 * 96) / 32, 64 / 32, B);
    conv4x4s2_wmma<3, 2><<<g, 32, 0, stream>>>(ybuf, wc1, c1_b, ping, nullptr,
        258, 194, 64, 128, 96, 130, 98, 64, 0, 1);
  }
  halo(ping, 130, 98, 64);
  {
    dim3 g((64 * 48) / 32, 128 / 32, B);
    conv4x4s2_wmma<6, 2><<<g, 32, 0, stream>>>(ping, wc2, c2_b, pong, nullptr,
        130, 98, 128, 64, 48, 66, 50, 128, 0, 1);
  }
  halo(pong, 66, 50, 128);
  {
    dim3 g((64 * 48) / 32, 64 / 32, B * 4);
    deconv4x4s2_wmma<7, 2><<<g, 32, 0, stream>>>(pong, wcd1, cd1_b, ping, nullptr,
        66, 50, 64, 64, 64, 48, 130, 98, 64, 0, 1);
  }
  halo(ping, 130, 98, 64);
  {
    dim3 g((128 * 96) / 32, 1, B * 4);
    deconv4x4s2_wmma<6, 1><<<g, 32, 0, stream>>>(ping, wcd2, cd2_b, nullptr,
        (float*)d_out, 130, 98, 3, 16, 128, 96, 0, 0, 0, 0, 3);
  }
}
